// RefineModel_90417651515965
// MI455X (gfx1250) — matrix operand order repaired and checked by reading, not compiled, not run on hardware
//
#include <hip/hip_runtime.h>

typedef __attribute__((ext_vector_type(16))) _Float16 v16h;
typedef __attribute__((ext_vector_type(8)))  float    v8f;

constexpr int kN  = 131072;
constexpr int kC  = 32;
constexpr int kK  = 19;
constexpr int kZD = 16;
constexpr float kBnEps  = 1e-4f;
constexpr float kSigThr = 0.8f;
constexpr float kRsC    = 0.17677669529663687f; // 1/sqrt(32)

// workspace layout (32-bit units; ws_u and ws_f alias the same buffer)
constexpr int WS_FRAGS = 0;            // 14 fragments * 256 u32 (pre-packed f16 B frags)
constexpr int WS_MK    = 14 * 256;     // 3*32 f32  : ws @ wk
constexpr int WS_MVO   = WS_MK + 96;   // 3*32 f32  : ws @ wv @ wo
constexpr int WS_BVOB  = WS_MVO + 96;  // 32   f32  : bv @ wo + bo

__device__ __forceinline__ unsigned int pack2h(float a, float b) {
    union { _Float16 h; unsigned short u; } x, y;
    x.h = (_Float16)a; y.h = (_Float16)b;
    return (unsigned int)x.u | ((unsigned int)y.u << 16);
}

// ---------------------------------------------------------------------------
// Kernel 1: fold constants and pre-pack weight B-fragments into WMMA lane
// layout.  B (32x16 f16 chunk): lane L -> column n = L&15, K rows
// 16*(L>>4) + {2j, 2j+1} in VGPR j.  Fragment stored lane-major:
// u32[lane*8 + j]  ->  kernel 2 loads each fragment as 2x b128 per lane.
// ---------------------------------------------------------------------------
__global__ __launch_bounds__(256) void precompute_kernel(
    const float* __restrict__ z,       const float* __restrict__ select_w1,
    const float* __restrict__ wq,      const float* __restrict__ wk,
    const float* __restrict__ wv,      const float* __restrict__ wo,
    const float* __restrict__ bv,      const float* __restrict__ bo,
    const float* __restrict__ wt,      const float* __restrict__ w_out,
    const float* __restrict__ w_gen,   const float* __restrict__ b_gen,
    float* __restrict__ ws_f, unsigned int* __restrict__ ws_u)
{
    __shared__ float s_base[kC];
    __shared__ float s_BK[kK * kC];
    __shared__ float s_BV[kK * kC];
    __shared__ float s_BVo[kK * kC];
    __shared__ float s_Mv[3 * kC];
    const int t = threadIdx.x;

    if (t < kC) {
        float acc = b_gen[t];
        for (int d = 0; d < kZD; ++d) acc += z[d] * w_gen[d * kC + t];
        s_base[t] = acc;
    }
    __syncthreads();

    // BK[k][c] = (base + wc[k]) @ wk ; BV likewise with wv
    for (int idx = t; idx < kK * kC; idx += blockDim.x) {
        int k = idx / kC, c = idx % kC;
        float a = 0.f, b = 0.f;
        for (int c1 = 0; c1 < kC; ++c1) {
            float g = s_base[c1] + w_gen[(kZD + k) * kC + c1];
            a += g * wk[c1 * kC + c];
            b += g * wv[c1 * kC + c];
        }
        s_BK[idx] = a;
        s_BV[idx] = b;
    }
    // Mk = ws@wk (to workspace), Mv = ws@wv (staged in LDS)
    for (int idx = t; idx < 3 * kC; idx += blockDim.x) {
        int d = idx / kC, c = idx % kC;
        float a = 0.f, b = 0.f;
        for (int e = 0; e < kC; ++e) {
            float wsw = w_gen[(kZD + kK + d) * kC + e];
            a += wsw * wk[e * kC + c];
            b += wsw * wv[e * kC + c];
        }
        ws_f[WS_MK + idx] = a;
        s_Mv[idx] = b;
    }
    __syncthreads();

    // fold wo:  BVo = BV@wo, Mvo = Mv@wo, bvo = bv@wo + bo
    for (int idx = t; idx < kK * kC; idx += blockDim.x) {
        int k = idx / kC, c = idx % kC;
        float a = 0.f;
        for (int c2 = 0; c2 < kC; ++c2) a += s_BV[k * kC + c2] * wo[c2 * kC + c];
        s_BVo[idx] = a;
    }
    for (int idx = t; idx < 3 * kC; idx += blockDim.x) {
        int d = idx / kC, c = idx % kC;
        float a = 0.f;
        for (int c2 = 0; c2 < kC; ++c2) a += s_Mv[d * kC + c2] * wo[c2 * kC + c];
        ws_f[WS_MVO + idx] = a;
    }
    if (t < kC) {
        float a = bo[t];
        for (int c2 = 0; c2 < kC; ++c2) a += bv[c2] * wo[c2 * kC + t];
        ws_f[WS_BVOB + t] = a;
    }
    __syncthreads();

    // pack 14 B fragments (lane-major):
    // 0/1 select_w1 | 2/3 wq | 4/5 BK^T (CxK, pad) | 6/7 BVo (pad K rows)
    // 8/9 wt[0:32] | 10/11 wt[32:64] | 12/13 w_out (32x19, pad cols)
    for (int idx = t; idx < 14 * 256; idx += blockDim.x) {
        int f    = idx >> 8;
        int rem  = idx & 255;
        int lane = rem >> 3;
        int j    = rem & 7;
        int n    = lane & 15;
        int k0   = ((lane >> 4) << 4) + 2 * j;
        float v[2];
        #pragma unroll
        for (int h = 0; h < 2; ++h) {
            int kk = k0 + h;
            float val;
            switch (f) {
            case 0: case 1:   val = select_w1[kk * kC + (f & 1) * 16 + n]; break;
            case 2: case 3:   val = wq[kk * kC + ((f - 2) & 1) * 16 + n]; break;
            case 4: case 5: { int cls = ((f - 4) & 1) * 16 + n;
                              val = (cls < kK) ? s_BK[cls * kC + kk] : 0.f; } break;
            case 6: case 7:   val = (kk < kK) ? s_BVo[kk * kC + ((f - 6) & 1) * 16 + n] : 0.f; break;
            case 8: case 9:   val = wt[kk * kC + ((f - 8) & 1) * 16 + n]; break;
            case 10: case 11: val = wt[(kC + kk) * kC + ((f - 10) & 1) * 16 + n]; break;
            default: { int col = ((f - 12) & 1) * 16 + n;
                       val = (col < kK) ? w_out[kk * kK + col] : 0.f; } break;
            }
            v[h] = val;
        }
        ws_u[WS_FRAGS + idx] = pack2h(v[0], v[1]);
    }
}

// ---------------------------------------------------------------------------
// helpers
// ---------------------------------------------------------------------------
__device__ __forceinline__ v8f wmma_f16(v16h a, v16h b, v8f c) {
    return __builtin_amdgcn_wmma_f32_16x16x32_f16(false, a, false, b,
                                                  (short)0, c, false, false);
}

// compiler-only memory fence: all LDS buffers are wave-private and DS ops
// from one wave execute in order (CDNA5 ISA 7.3), so no s_barrier is needed.
__device__ __forceinline__ void wave_fence() { asm volatile("" ::: "memory"); }

// LDS aperture address = addr[31:0] (CDNA5 ISA 10.2)
__device__ __forceinline__ unsigned lds_addr(const void* p) {
    return (unsigned)(unsigned long long)p;
}

// async global -> LDS, 128-bit, saddr form (tracked by ASYNCcnt)
__device__ __forceinline__ void async_load_b128(unsigned lds, unsigned voff,
                                                unsigned long long gbase) {
    asm volatile("global_load_async_to_lds_b128 %0, %1, %2"
                 :: "v"(lds), "v"(voff), "s"(gbase) : "memory");
}
// async LDS -> global, 128-bit
__device__ __forceinline__ void async_store_b128(unsigned voff, unsigned lds,
                                                 unsigned long long gbase) {
    asm volatile("global_store_async_from_lds_b128 %0, %1, %2"
                 :: "v"(voff), "v"(lds), "s"(gbase) : "memory");
}
__device__ __forceinline__ void wait_async0() {
    asm volatile("s_wait_asynccnt 0" ::: "memory");
}

// A 16x32 f16: lane m = L&15; K = base(j,h) + 16*(L>=16),
// base(j,h) = 2j+h
__device__ __forceinline__ v16h packA(const float* __restrict__ buf, int lane, int ncols) {
    int m    = lane & 15;
    int koff = (lane & 16) ? 16 : 0;
    v16h a;
    #pragma unroll
    for (int j = 0; j < 8; ++j) {
        int kb = 2 * j;
        #pragma unroll
        for (int h = 0; h < 2; ++h) {
            int kk = kb + h + koff;
            float v = (kk < ncols) ? buf[m * kC + kk] : 0.0f;
            a[2 * j + h] = (_Float16)v;
        }
    }
    return a;
}

__device__ __forceinline__ v16h loadB(const unsigned int* __restrict__ frag, int lane) {
    union { v16h h; uint4 u[2]; } r;
    const uint4* p = (const uint4*)frag + lane * 2;
    r.u[0] = p[0];
    r.u[1] = p[1];
    return r.h;
}

// ---------------------------------------------------------------------------
// Kernel 2: main pipeline.  8 waves/block, 16 rows/wave, 128 rows/block.
// All LDS buffers are wave-private; waves run fully decoupled (no barriers).
// ---------------------------------------------------------------------------
__global__ __launch_bounds__(256) void refine_kernel(
    const float* __restrict__ feat,     const float* __restrict__ scene_flow,
    const float* __restrict__ coarse_pred,
    const float* __restrict__ bn_gamma, const float* __restrict__ bn_beta,
    const float* __restrict__ bn_mean,  const float* __restrict__ bn_var,
    const float* __restrict__ select_w2,const float* __restrict__ select_b2,
    const float* __restrict__ bq,       const float* __restrict__ bk,
    const float* __restrict__ bt,       const float* __restrict__ b_out,
    const float* __restrict__ ws_f,     const unsigned int* __restrict__ ws_u,
    float* __restrict__ out)
{
    __shared__ __align__(16) float s_buf[8][16 * kC];   // per-wave staging tile
    __shared__ __align__(16) float s_sf [8][16 * 3];    // scene_flow tile
    __shared__ __align__(16) float s_cp [8][16 * kK];   // softmaxed coarse preds
    __shared__ float s_sc [8][16];        // s = sum(attn*cp)
    __shared__ float s_qsf[8][16];        // q . sf_k
    __shared__ float s_qb [8][16];        // q . bk
    __shared__ float s_vm [8][16];        // valid mask (0/1)

    const int w       = threadIdx.x >> 5;
    const int lane    = threadIdx.x & 31;
    const int rowbase = blockIdx.x * 128 + w * 16;
    float* buf = s_buf[w];
    float* sfb = s_sf[w];
    float* cpb = s_cp[w];

    // ---- stage inputs: async global->LDS b128 (ASYNCcnt) ---------------
    {
        const unsigned bufl = lds_addr(buf);
        const unsigned cpl  = lds_addr(cpb);
        const unsigned sfl  = lds_addr(sfb);
        const unsigned long long gfeat =
            (unsigned long long)(feat + (size_t)rowbase * kC);     // 2048 B
        const unsigned long long gcp =
            (unsigned long long)(coarse_pred + (size_t)rowbase * kK); // 1216 B
        const unsigned long long gsf =
            (unsigned long long)(scene_flow + (size_t)rowbase * 3);   // 192 B
        #pragma unroll
        for (int i = 0; i < 4; ++i) {
            unsigned off = (unsigned)(i * 32 + lane) * 16u;
            async_load_b128(bufl + off, off, gfeat);
        }
        for (int idx = lane; idx < 76; idx += 32) {
            unsigned off = (unsigned)idx * 16u;
            async_load_b128(cpl + off, off, gcp);
        }
        if (lane < 12) {
            unsigned off = (unsigned)lane * 16u;
            async_load_b128(sfl + off, off, gsf);
        }
        wait_async0();
    }
    wave_fence();

    // ---- per-row: softmax(cp), temporal mask ---------------------------
    if (lane < 16) {
        float mx = -1e30f;
        #pragma unroll
        for (int k = 0; k < kK; ++k) mx = fmaxf(mx, cpb[lane * kK + k]);
        float e[kK], sum = 0.f;
        #pragma unroll
        for (int k = 0; k < kK; ++k) { e[k] = __expf(cpb[lane * kK + k] - mx); sum += e[k]; }
        float inv = 1.f / sum;
        #pragma unroll
        for (int k = 0; k < kK; ++k) cpb[lane * kK + k] = e[k] * inv;
        float m = ((sfb[lane * 3 + 0] != 0.f) || (sfb[lane * 3 + 1] != 0.f) ||
                   (sfb[lane * 3 + 2] != 0.f)) ? 1.f : 0.f;
        s_vm[w][lane] = m;
    }
    wave_fence();

    const v16h afF = packA(buf, lane, kC);     // feat A-frag (reused 2x)
    const v8f  z8  = {};
    wave_fence();

    // ---- select branch: h = relu(BN(feat@W1)); score = sigmoid(h@w2+b2)
    {
        v16h b0 = loadB(ws_u + 0 * 256, lane), b1 = loadB(ws_u + 1 * 256, lane);
        v8f  d0 = wmma_f16(afF, b0, z8),       d1 = wmma_f16(afF, b1, z8);
        #pragma unroll
        for (int tile = 0; tile < 2; ++tile) {
            v8f d = tile ? d1 : d0;
            int col = (lane & 15) + tile * 16;
            float sc = bn_gamma[col] * rsqrtf(bn_var[col] + kBnEps);
            float mu = bn_mean[col], be = bn_beta[col], w2 = select_w2[col];
            int rb = (lane >> 4) * 8;
            #pragma unroll
            for (int r = 0; r < 8; ++r) {
                float h = (d[r] - mu) * sc + be;
                buf[(rb + r) * kC + col] = fmaxf(h, 0.f) * w2;
            }
        }
    }
    wave_fence();
    if (lane < 16) {
        float a = select_b2[0];
        #pragma unroll
        for (int c = 0; c < kC; ++c) a += buf[lane * kC + c];
        float score = 1.f / (1.f + __expf(-a));
        s_vm[w][lane] = (score < kSigThr) ? s_vm[w][lane] : 0.f;
    }
    wave_fence();

    // ---- q = feat@wq + bq ---------------------------------------------
    {
        v16h b0 = loadB(ws_u + 2 * 256, lane), b1 = loadB(ws_u + 3 * 256, lane);
        v8f  d0 = wmma_f16(afF, b0, z8),       d1 = wmma_f16(afF, b1, z8);
        #pragma unroll
        for (int tile = 0; tile < 2; ++tile) {
            v8f d = tile ? d1 : d0;
            int col = (lane & 15) + tile * 16;
            float bqc = bq[col];
            int rb = (lane >> 4) * 8;
            #pragma unroll
            for (int r = 0; r < 8; ++r) buf[(rb + r) * kC + col] = d[r] + bqc;
        }
    }
    wave_fence();

    // ---- per-row: qsf = q.(sf@Mk), qb = q.bk ---------------------------
    if (lane < 16) {
        const float* q = buf + lane * kC;
        float t0 = 0.f, t1 = 0.f, t2 = 0.f, qbv = 0.f;
        #pragma unroll
        for (int c = 0; c < kC; ++c) {
            float qc = q[c];
            t0  += ws_f[WS_MK + c] * qc;
            t1  += ws_f[WS_MK + 32 + c] * qc;
            t2  += ws_f[WS_MK + 64 + c] * qc;
            qbv += bk[c] * qc;
        }
        s_qsf[w][lane] = sfb[lane * 3] * t0 + sfb[lane * 3 + 1] * t1 + sfb[lane * 3 + 2] * t2;
        s_qb[w][lane]  = qbv;
    }
    const v16h afQ = packA(buf, lane, kC);
    wave_fence();

    // ---- logits base: q @ BK^T  (N x 19, padded) -----------------------
    {
        v16h b0 = loadB(ws_u + 4 * 256, lane), b1 = loadB(ws_u + 5 * 256, lane);
        v8f  d0 = wmma_f16(afQ, b0, z8),       d1 = wmma_f16(afQ, b1, z8);
        #pragma unroll
        for (int tile = 0; tile < 2; ++tile) {
            v8f d = tile ? d1 : d0;
            int col = (lane & 15) + tile * 16;
            int rb = (lane >> 4) * 8;
            #pragma unroll
            for (int r = 0; r < 8; ++r) buf[(rb + r) * kC + col] = d[r];
        }
    }
    wave_fence();

    // ---- per-row: attention softmax, w = attn*cp, s = sum(w) -----------
    if (lane < 16) {
        float qsf = s_qsf[w][lane], qbv = s_qb[w][lane];
        float l[kK], mx = -1e30f;
        #pragma unroll
        for (int k = 0; k < kK; ++k) {
            float cpv = cpb[lane * kK + k];
            l[k] = (cpv * (buf[lane * kC + k] + qsf) + qbv) * kRsC;
            mx = fmaxf(mx, l[k]);
        }
        float sum = 0.f;
        #pragma unroll
        for (int k = 0; k < kK; ++k) { l[k] = __expf(l[k] - mx); sum += l[k]; }
        float inv = 1.f / sum, ssum = 0.f;
        #pragma unroll
        for (int k = 0; k < kK; ++k) {
            float wv = l[k] * inv * cpb[lane * kK + k];
            buf[lane * kC + k] = wv;
            ssum += wv;
        }
        s_sc[w][lane] = ssum;
    }
    wave_fence();
    const v16h afW = packA(buf, lane, kK);   // zero-padded K -> 32
    wave_fence();

    // ---- aug = w@BVo + s*(sf@Mvo) + bvo; new_feat = vm ? aug : 0 -------
    {
        v16h b0 = loadB(ws_u + 6 * 256, lane), b1 = loadB(ws_u + 7 * 256, lane);
        v8f  d0 = wmma_f16(afW, b0, z8),       d1 = wmma_f16(afW, b1, z8);
        #pragma unroll
        for (int tile = 0; tile < 2; ++tile) {
            v8f d = tile ? d1 : d0;
            int col = (lane & 15) + tile * 16;
            float m0 = ws_f[WS_MVO + col], m1 = ws_f[WS_MVO + 32 + col],
                  m2 = ws_f[WS_MVO + 64 + col], bb = ws_f[WS_BVOB + col];
            int rb = (lane >> 4) * 8;
            #pragma unroll
            for (int r = 0; r < 8; ++r) {
                int row = rb + r;
                float sfv = sfb[row * 3] * m0 + sfb[row * 3 + 1] * m1 + sfb[row * 3 + 2] * m2;
                float aug = d[r] + s_sc[w][row] * sfv + bb;
                buf[row * kC + col] = s_vm[w][row] * aug;
            }
        }
    }
    wave_fence();
    const v16h afNF = packA(buf, lane, kC);
    wave_fence();

    // ---- refine = feat@wt_top + new_feat@wt_bot + bt (C-operand chain) -
    {
        v16h b0 = loadB(ws_u + 8 * 256, lane),  v16h_b1 = loadB(ws_u + 9 * 256, lane);
        v8f  d0 = wmma_f16(afF, b0, z8),        d1 = wmma_f16(afF, v16h_b1, z8);
        v16h b2 = loadB(ws_u + 10 * 256, lane), b3 = loadB(ws_u + 11 * 256, lane);
        d0 = wmma_f16(afNF, b2, d0);
        d1 = wmma_f16(afNF, b3, d1);
        #pragma unroll
        for (int tile = 0; tile < 2; ++tile) {
            v8f d = tile ? d1 : d0;
            int col = (lane & 15) + tile * 16;
            float btc = bt[col];
            int rb = (lane >> 4) * 8;
            #pragma unroll
            for (int r = 0; r < 8; ++r) buf[(rb + r) * kC + col] = d[r] + btc;
        }
    }
    wave_fence();
    const v16h afR = packA(buf, lane, kC);
    wave_fence();

    // ---- out = refine @ w_out + b_out; async LDS->global store ---------
    {
        v16h b0 = loadB(ws_u + 12 * 256, lane), b1 = loadB(ws_u + 13 * 256, lane);
        v8f  d0 = wmma_f16(afR, b0, z8),        d1 = wmma_f16(afR, b1, z8);
        #pragma unroll
        for (int tile = 0; tile < 2; ++tile) {
            v8f d = tile ? d1 : d0;
            int col = (lane & 15) + tile * 16;
            if (col < kK) {
                float bc = b_out[col];
                int rb = (lane >> 4) * 8;
                #pragma unroll
                for (int r = 0; r < 8; ++r)
                    buf[(rb + r) * kK + col] = d[r] + bc;   // contiguous 16x19
            }
        }
        wave_fence();
        const unsigned bufl = lds_addr(buf);
        const unsigned long long gout =
            (unsigned long long)(out + (size_t)rowbase * kK);   // 1216 B, 16B-aligned
        for (int idx = lane; idx < 76; idx += 32) {
            unsigned off = (unsigned)idx * 16u;
            async_store_b128(off, bufl + off, gout);
        }
        wait_async0();
    }
}

// ---------------------------------------------------------------------------
extern "C" void kernel_launch(void* const* d_in, const int* in_sizes, int n_in,
                              void* d_out, int out_size, void* d_ws, size_t ws_size,
                              hipStream_t stream) {
    const float* feat        = (const float*)d_in[0];
    const float* scene_flow  = (const float*)d_in[1];
    const float* coarse_pred = (const float*)d_in[2];
    const float* z           = (const float*)d_in[3];
    const float* select_w1   = (const float*)d_in[4];
    const float* bn_gamma    = (const float*)d_in[5];
    const float* bn_beta     = (const float*)d_in[6];
    const float* bn_mean     = (const float*)d_in[7];
    const float* bn_var      = (const float*)d_in[8];
    const float* select_w2   = (const float*)d_in[9];
    const float* select_b2   = (const float*)d_in[10];
    const float* wq          = (const float*)d_in[11];
    const float* bq          = (const float*)d_in[12];
    const float* wk          = (const float*)d_in[13];
    const float* bk          = (const float*)d_in[14];
    const float* wv          = (const float*)d_in[15];
    const float* bv          = (const float*)d_in[16];
    const float* wo          = (const float*)d_in[17];
    const float* bo          = (const float*)d_in[18];
    const float* wt          = (const float*)d_in[19];
    const float* bt          = (const float*)d_in[20];
    const float* w_out       = (const float*)d_in[21];
    const float* b_out       = (const float*)d_in[22];
    const float* w_gen       = (const float*)d_in[23];
    const float* b_gen       = (const float*)d_in[24];

    float*        wsf = (float*)d_ws;
    unsigned int* wsu = (unsigned int*)d_ws;

    hipLaunchKernelGGL(precompute_kernel, dim3(1), dim3(256), 0, stream,
                       z, select_w1, wq, wk, wv, wo, bv, bo, wt, w_out,
                       w_gen, b_gen, wsf, wsu);

    hipLaunchKernelGGL(refine_kernel, dim3(kN / 128), dim3(256), 0, stream,
                       feat, scene_flow, coarse_pred,
                       bn_gamma, bn_beta, bn_mean, bn_var,
                       select_w2, select_b2, bq, bk, bt, b_out,
                       wsf, wsu, (float*)d_out);
}
